// AttDecoder_78907139162633
// MI455X (gfx1250) — matrix operand order repaired and checked by reading, not compiled, not run on hardware
//
#include <hip/hip_runtime.h>

// ---------------------------------------------------------------------------
// AttDecoder for MI455X (gfx1250).
// GEMMs: bf16 WMMA (v_wmma_f32_16x16x32_bf16), f32 accumulate,
//        TDM (tensor_load_to_lds) double-buffered tile staging.
// Attention/softmax/LSTM pointwise: fp32 VALU (<1% of FLOPs).
// ---------------------------------------------------------------------------

typedef __bf16 bf16_t;
typedef __attribute__((ext_vector_type(16))) __bf16 v16bf;
typedef __attribute__((ext_vector_type(8)))  float  v8f;
typedef __attribute__((ext_vector_type(4)))  unsigned u32x4_t;
typedef __attribute__((ext_vector_type(8)))  int      i32x8_t;
typedef __attribute__((ext_vector_type(4)))  int      i32x4_t;

namespace {
constexpr int kB = 64, kS = 128, kC = 512, kH = 1024, kE = 512, kV = 32000, kT = 16;
constexpr int kG = 4 * kH;       // 4096 gate width
constexpr int kF = kE + kH;      // 1536 feats width
constexpr int kBS = kB * kS;     // 8192 context rows
}

union FragBF { v16bf v; unsigned u[8]; uint4 q[2]; };

// ---------------------------------------------------------------------------
// TDM: issue a 2D bf16 tile load (rows x rowLen elems, row stride strideElems)
// from global `src` into LDS byte offset `ldsByteOff`. Descriptor packing per
// CDNA5 ISA (D# group0 + group1; groups 2/3 zero = 2D).
// data_size=1 (2-byte elements); tile stored compacted row-major in LDS.
// Toolchain uses the 6-arg builtin: (u32x4, i32x8, i32x4, i32x4, i32x8, cpol).
// ---------------------------------------------------------------------------
__device__ __forceinline__ void tdm_load_tile_bf16(const bf16_t* src,
                                                   unsigned ldsByteOff,
                                                   int rows, int rowLenElems,
                                                   long strideElems) {
  const unsigned long long ga = (unsigned long long)(size_t)src;
  u32x4_t g0;
  g0[0] = 1u;                                        // count=1, user mode
  g0[1] = ldsByteOff;                                // lds_addr [63:32]
  g0[2] = (unsigned)(ga & 0xFFFFFFFFull);            // global_addr lo
  g0[3] = (unsigned)((ga >> 32) & 0x1FFFFFFull)      // global_addr hi (57b)
          | 0x80000000u;                             // type=2 ("image")
  const unsigned td0 = (unsigned)strideElems;        // tensor_dim0 (OOB-benign)
  const unsigned td1 = 1u << 20;                     // tensor_dim1 (OOB-benign)
  const unsigned long long st0 = (unsigned long long)strideElems;
  i32x8_t g1;
  g1[0] = (int)(1u << 16);                                           // data_size=1 (2B)
  g1[1] = (int)((td0 & 0xFFFFu) << 16);                              // tensor_dim0 lo16
  g1[2] = (int)(((td0 >> 16) & 0xFFFFu) | ((td1 & 0xFFFFu) << 16));  // dim0 hi | dim1 lo
  g1[3] = (int)(((td1 >> 16) & 0xFFFFu)
          | (((unsigned)rowLenElems & 0xFFFFu) << 16));              // dim1 hi | tile_dim0
  g1[4] = (int)((unsigned)rows & 0xFFFFu);                           // tile_dim1; tile_dim2=0
  g1[5] = (int)(unsigned)(st0 & 0xFFFFFFFFull);                      // dim0_stride lo32
  g1[6] = (int)(unsigned)((st0 >> 32) & 0xFFFFull);                  // dim0_stride hi16
  g1[7] = 0;                                                         // dim1_stride (unused, 2D)
  const i32x4_t z4 = {0, 0, 0, 0};
  const i32x8_t z8 = {0, 0, 0, 0, 0, 0, 0, 0};
  __builtin_amdgcn_tensor_load_to_lds(g0, g1, z4, z4, z8, 0);
}

// ---------------------------------------------------------------------------
// bf16 WMMA GEMM:  C[M,N] = A0[M,K0]*B0[N,K0]^T (+ A1[M,K1]*B1[N,K1]^T if DUAL)
//                  + bias0 (+ bias1 if DUAL), written f32 or bf16.
// Block = 256 threads = 8 waves; block tile = 32(M) x 256(N);
// waves arranged 2(M) x 4(N); each wave computes 16x64 (4 WMMA tiles) reusing
// one A fragment. Tiles staged by TDM with double buffering: the DMA for
// chunk i+1 overlaps WMMA on chunk i; wave0 owns TENSORcnt, a workgroup
// barrier publishes each tile. One barrier per K-chunk. All fragments are
// preloaded as 16B ds_load_b128 before the 4 back-to-back WMMAs.
// ---------------------------------------------------------------------------
template <bool DUAL, bool OUT_BF16>
__global__ __launch_bounds__(256)
void wmma_gemm_kernel(const bf16_t* __restrict__ A0, int lda0,
                      const bf16_t* __restrict__ B0, int ldb0, int K0,
                      const bf16_t* __restrict__ A1, int lda1,
                      const bf16_t* __restrict__ B1, int ldb1, int K1,
                      const float* __restrict__ bias0, const float* __restrict__ bias1,
                      float* __restrict__ Cf, bf16_t* __restrict__ Cb,
                      long ldc, long coff) {
  // Single static LDS block: A tiles at dword [0..1023] (2 bufs x 32x32 bf16),
  // B tiles at dword [1024..9215] (2 bufs x 256x32 bf16). 36 KB total.
  __shared__ __align__(16) unsigned smem[2 * 32 * 16 + 2 * 256 * 16];

  const int tid   = threadIdx.x;
  const int wave  = tid >> 5;
  const int lane  = tid & 31;
  const int waveM = wave >> 2;        // 0..1
  const int waveN = wave & 3;         // 0..3
  const int laneN = lane & 15;
  const int hi    = lane >> 4;        // 0 / 1
  const int m0    = blockIdx.y * 32;
  const int n0    = blockIdx.x * 256;

  const int nc0   = K0 >> 5;
  const int total = nc0 + (DUAL ? (K1 >> 5) : 0);

  v8f acc[4] = {};

  auto issue = [&](int c, int buf) {
    const bf16_t* Ap; const bf16_t* Bp; int lda, ldb, k;
    if (!DUAL || c < nc0) { Ap = A0; Bp = B0; lda = lda0; ldb = ldb0; k = c << 5; }
    else                  { Ap = A1; Bp = B1; lda = lda1; ldb = ldb1; k = (c - nc0) << 5; }
    tdm_load_tile_bf16(Ap + (size_t)m0 * lda + k, (unsigned)(buf * 2048),          32,  32, lda);
    tdm_load_tile_bf16(Bp + (size_t)n0 * ldb + k, (unsigned)(4096 + buf * 16384), 256,  32, ldb);
  };

  if (wave == 0) issue(0, 0);

  for (int c = 0; c < total; ++c) {
    const int buf = c & 1;
    if (wave == 0) __builtin_amdgcn_s_wait_tensorcnt((short)0);
    __syncthreads();                                  // tile `buf` is ready for all waves
    if (wave == 0 && (c + 1) < total) issue(c + 1, buf ^ 1);

    const unsigned aBase = (unsigned)(buf * 512);
    const unsigned bBase = (unsigned)(1024 + buf * 4096);

    // A fragment (16-bit A 16x32, contiguous k): lanes0-15 K0..15,
    // lanes16-31 K16..31. 8 contiguous dwords -> 2x ds_load_b128.
    FragBF a;
    const unsigned aOff = aBase + (unsigned)(waveM * 16 + laneN) * 16 + (unsigned)(hi * 8);
    a.q[0] = *(const uint4*)&smem[aOff];
    a.q[1] = *(const uint4*)&smem[aOff + 4];

    // B fragments (32x16 layout): lane column laneN; lanes0-15 K0..15,
    // lanes16-31 K16..31; 8 contiguous dwords -> 2x ds_load_b128 each.
    // Preload all 4 so the 4 WMMAs issue back-to-back after one DS wait.
    FragBF bfr[4];
    #pragma unroll
    for (int t = 0; t < 4; ++t) {
      const unsigned bOff = bBase + (unsigned)(waveN * 64 + t * 16 + laneN) * 16
                          + (unsigned)(hi * 8);
      bfr[t].q[0] = *(const uint4*)&smem[bOff];
      bfr[t].q[1] = *(const uint4*)&smem[bOff + 4];
    }
    #pragma unroll
    for (int t = 0; t < 4; ++t) {
      acc[t] = __builtin_amdgcn_wmma_f32_16x16x32_bf16(false, a.v, false, bfr[t].v,
                                                       (short)0, acc[t], false, false);
    }
    // Next iteration's leading barrier orders LDS reuse against TDM writes.
  }

  // Epilogue: C layout — VGPR r holds M=r (lanes 0-15) / M=r+8 (lanes 16-31).
  #pragma unroll
  for (int t = 0; t < 4; ++t) {
    const int nc = n0 + waveN * 64 + t * 16 + laneN;
    float bsum = bias0[nc];
    if (DUAL) bsum += bias1[nc];
    #pragma unroll
    for (int r = 0; r < 8; ++r) {
      const int m = m0 + waveM * 16 + r + hi * 8;
      const size_t off = (size_t)m * ldc + coff + nc;
      const float val = acc[t][r] + bsum;
      if (OUT_BF16) Cb[off] = (bf16_t)val;
      else          Cf[off] = val;
    }
  }
}

// ---------------------------------------------------------------------------
// Attention step: scores = key[b] @ h[b], softmax over S, pooled = att @ ctx[b].
// One block per batch row; fp32 VALU. Writes pooled as bf16 for the Wv GEMM.
// ---------------------------------------------------------------------------
__global__ __launch_bounds__(256)
void attention_kernel(const float* __restrict__ key, const float* __restrict__ ctx,
                      const float* __restrict__ h, bf16_t* __restrict__ pooledB) {
  __shared__ float sh[kH];
  __shared__ float satt[kS];
  __shared__ float sred[256];
  const int b = blockIdx.x, tid = threadIdx.x;

  for (int i = tid; i < kH; i += 256) sh[i] = h[(size_t)b * kH + i];
  __syncthreads();

  const int s = tid >> 1, part = tid & 1;
  const float* krow = key + ((size_t)b * kS + s) * kH + part * (kH / 2);
  float acc = 0.f;
  for (int k = 0; k < kH / 2; ++k) acc += krow[k] * sh[part * (kH / 2) + k];
  sred[tid] = acc;
  __syncthreads();
  if (part == 0) satt[s] = sred[tid] + sred[tid + 1];
  __syncthreads();

  sred[tid] = (tid < kS) ? satt[tid] : -3.402823466e38f;
  __syncthreads();
  for (int off = 128; off >= 1; off >>= 1) {
    if (tid < off) sred[tid] = fmaxf(sred[tid], sred[tid + off]);
    __syncthreads();
  }
  const float mx = sred[0];
  __syncthreads();
  float e = 0.f;
  if (tid < kS) e = __expf(satt[tid] - mx);
  sred[tid] = e;
  __syncthreads();
  for (int off = 128; off >= 1; off >>= 1) {
    if (tid < off) sred[tid] += sred[tid + off];
    __syncthreads();
  }
  const float inv = 1.f / sred[0];
  __syncthreads();
  if (tid < kS) satt[tid] = e * inv;
  __syncthreads();

  float p0 = 0.f, p1 = 0.f;
  for (int sI = 0; sI < kS; ++sI) {
    const float a = satt[sI];
    const float* crow = ctx + ((size_t)b * kS + sI) * kC;
    p0 += a * crow[tid];
    p1 += a * crow[tid + 256];
  }
  pooledB[(size_t)b * kC + tid]       = (bf16_t)p0;
  pooledB[(size_t)b * kC + tid + 256] = (bf16_t)p1;
}

// ---------------------------------------------------------------------------
// Small helper kernels
// ---------------------------------------------------------------------------
__global__ void cvt_kernel(const float* __restrict__ src, bf16_t* __restrict__ dst, int n) {
  const int i = blockIdx.x * 256 + threadIdx.x;
  if (i < n) dst[i] = (bf16_t)src[i];
}

__global__ void init_state_kernel(const float* __restrict__ h0, const float* __restrict__ c0,
                                  float* __restrict__ h, float* __restrict__ c,
                                  bf16_t* __restrict__ hB) {
  const int i = blockIdx.x * 256 + threadIdx.x;   // B*H
  const float hv = h0[i];
  h[i] = hv; c[i] = c0[i]; hB[i] = (bf16_t)hv;
}

__global__ void embed_gather_kernel(const float* __restrict__ embed, const int* __restrict__ data,
                                    int t, bf16_t* __restrict__ feats) {
  const int i = blockIdx.x * 256 + threadIdx.x;   // B*E
  const int b = i >> 9, e = i & 511;
  const int tok = data[b * kT + t];
  feats[(size_t)b * kF + e] = (bf16_t)embed[(size_t)tok * kE + e];
}

__device__ __forceinline__ float sigmoidf_(float x) { return 1.f / (1.f + __expf(-x)); }

__global__ void lstm_kernel(const float* __restrict__ gates, float* __restrict__ h,
                            float* __restrict__ c, bf16_t* __restrict__ hB) {
  const int i = blockIdx.x * 256 + threadIdx.x;   // B*H
  const int b = i >> 10, j = i & 1023;
  const float* g = gates + (size_t)b * kG;
  const float gi = g[j], gf = g[kH + j], gg = g[2 * kH + j], go = g[3 * kH + j];
  const float cn = sigmoidf_(gf) * c[i] + sigmoidf_(gi) * tanhf(gg);
  const float hn = sigmoidf_(go) * tanhf(cn);
  c[i] = cn; h[i] = hn; hB[i] = (bf16_t)hn;
}

__global__ void tail_kernel(const float* __restrict__ h, const float* __restrict__ c,
                            float* __restrict__ out) {
  const int i = blockIdx.x * 256 + threadIdx.x;   // B*H
  const size_t base = (size_t)kB * kT * kV;
  out[base + i] = h[i];
  out[base + (size_t)kB * kH + i] = c[i];
}

// ---------------------------------------------------------------------------
// Host launcher
// ---------------------------------------------------------------------------
extern "C" void kernel_launch(void* const* d_in, const int* in_sizes, int n_in,
                              void* d_out, int out_size, void* d_ws, size_t ws_size,
                              hipStream_t stream) {
  (void)in_sizes; (void)n_in; (void)out_size; (void)ws_size;

  const float* context = (const float*)d_in[0];
  const int*   data    = (const int*)  d_in[1];
  const float* h0      = (const float*)d_in[2];
  const float* c0      = (const float*)d_in[3];
  const float* Wk      = (const float*)d_in[4];
  const float* bk      = (const float*)d_in[5];
  const float* Wv      = (const float*)d_in[6];
  const float* bv      = (const float*)d_in[7];
  const float* embed   = (const float*)d_in[8];
  const float* W_ih    = (const float*)d_in[9];
  const float* W_hh    = (const float*)d_in[10];
  const float* b_ih    = (const float*)d_in[11];
  const float* b_hh    = (const float*)d_in[12];
  const float* Wp      = (const float*)d_in[13];
  const float* bp      = (const float*)d_in[14];
  float* out = (float*)d_out;

  // Workspace carve-out (256B aligned). Total ~133 MB.
  char* p = (char*)d_ws;
  auto alloc = [&](size_t bytes) -> void* {
    void* r = (void*)p;
    p += (bytes + 255) & ~(size_t)255;
    return r;
  };
  bf16_t* WkB     = (bf16_t*)alloc((size_t)kH * kC * 2);
  bf16_t* WvB     = (bf16_t*)alloc((size_t)kH * kC * 2);
  bf16_t* WihB    = (bf16_t*)alloc((size_t)kG * kF * 2);
  bf16_t* WhhB    = (bf16_t*)alloc((size_t)kG * kH * 2);
  bf16_t* WpB     = (bf16_t*)alloc((size_t)kV * kH * 2);
  bf16_t* ctxB    = (bf16_t*)alloc((size_t)kBS * kC * 2);
  float*  keyf    = (float*) alloc((size_t)kBS * kH * 4);
  bf16_t* featsB  = (bf16_t*)alloc((size_t)kB * kF * 2);
  bf16_t* pooledB = (bf16_t*)alloc((size_t)kB * kC * 2);
  bf16_t* hB      = (bf16_t*)alloc((size_t)kB * kH * 2);
  float*  hf      = (float*) alloc((size_t)kB * kH * 4);
  float*  cf      = (float*) alloc((size_t)kB * kH * 4);
  float*  gates   = (float*) alloc((size_t)kB * kG * 4);

  auto cvt = [&](const float* s, bf16_t* d, size_t n) {
    cvt_kernel<<<dim3((unsigned)((n + 255) / 256)), dim3(256), 0, stream>>>(s, d, (int)n);
  };
  cvt(Wk,   WkB,  (size_t)kH * kC);
  cvt(Wv,   WvB,  (size_t)kH * kC);
  cvt(W_ih, WihB, (size_t)kG * kF);
  cvt(W_hh, WhhB, (size_t)kG * kH);
  cvt(Wp,   WpB,  (size_t)kV * kH);
  cvt(context, ctxB, (size_t)kBS * kC);
  init_state_kernel<<<kB * kH / 256, 256, 0, stream>>>(h0, c0, hf, cf, hB);

  // key = context @ Wk^T + bk   [8192 x 1024], K=512
  wmma_gemm_kernel<false, false><<<dim3(kH / 256, kBS / 32), 256, 0, stream>>>(
      ctxB, kC, WkB, kC, kC,
      nullptr, 0, nullptr, 0, 0,
      bk, nullptr, keyf, nullptr, (long)kH, 0);

  for (int t = 0; t < kT; ++t) {
    attention_kernel<<<kB, 256, 0, stream>>>(keyf, context, hf, pooledB);

    // val = pooled @ Wv^T + bv  -> feats[:, E:E+H] (bf16)
    wmma_gemm_kernel<false, true><<<dim3(kH / 256, kB / 32), 256, 0, stream>>>(
        pooledB, kC, WvB, kC, kC,
        nullptr, 0, nullptr, 0, 0,
        bv, nullptr, nullptr, featsB, (long)kF, (long)kE);

    embed_gather_kernel<<<kB * kE / 256, 256, 0, stream>>>(embed, data, t, featsB);

    // gates = feats @ W_ih^T + b_ih + h @ W_hh^T + b_hh   [64 x 4096]
    wmma_gemm_kernel<true, false><<<dim3(kG / 256, kB / 32), 256, 0, stream>>>(
        featsB, kF, WihB, kF, kF,
        hB, kH, WhhB, kH, kH,
        b_ih, b_hh, gates, nullptr, (long)kG, 0);

    lstm_kernel<<<kB * kH / 256, 256, 0, stream>>>(gates, hf, cf, hB);

    // pred = h_new @ Wp^T + bp  -> out[:, t, :]  (row stride T*V)
    wmma_gemm_kernel<false, false><<<dim3(kV / 256, kB / 32), 256, 0, stream>>>(
        hB, kH, WpB, kH, kH,
        nullptr, 0, nullptr, 0, 0,
        bp, nullptr, out + (size_t)t * kV, nullptr, (long)kT * kV, 0);
  }

  tail_kernel<<<kB * kH / 256, 256, 0, stream>>>(hf, cf, out);
}
